// MultiHeadAttention_26388279067067
// MI455X (gfx1250) — matrix operand order repaired and checked by reading, not compiled, not run on hardware
//
#include <hip/hip_runtime.h>
#include <hip/hip_bf16.h>
#include <stdint.h>

#define B_  2
#define S_  2048
#define D_  768
#define H_  12
#define HD_ 64
#define BS_ (B_*S_)
#define KB_ 128   // K-step staged per TDM in the projection GEMM

typedef __attribute__((ext_vector_type(16))) __bf16 bfrag;   // WMMA A/B fragment
typedef __attribute__((ext_vector_type(8)))  float  v8f;     // WMMA C/D fragment
typedef __attribute__((ext_vector_type(4)))  unsigned int u32x4;
typedef __attribute__((ext_vector_type(8)))  int i32x8;
typedef __attribute__((ext_vector_type(4)))  int i32x4;

#if __has_builtin(__builtin_amdgcn_tensor_load_to_lds) && __has_builtin(__builtin_amdgcn_s_wait_tensorcnt)
#define USE_TDM 1
#else
#define USE_TDM 0
#endif

union FragU { bfrag f; u32x4 u[2]; };

__device__ __forceinline__ unsigned short f2bf(float f) {
  unsigned int u = __float_as_uint(f);
  unsigned int r = (u + 0x7FFFu + ((u >> 16) & 1u)) >> 16;   // RNE
  return (unsigned short)r;
}

// A-matrix 16x32 bf16 fragment from row-major matrix.
// lane<16: row=lane, K 0..7 & 16..23 ; lane>=16: row=lane-16, K 8..15 & 24..31
__device__ __forceinline__ bfrag load_a_frag(const unsigned short* base, int ld, int lane) {
  const int m    = lane & 15;
  const int koff = (lane < 16) ? 0 : 8;
  const unsigned short* p = base + (size_t)m * ld + koff;
  FragU r;
  r.u[0] = *(const u32x4*)(p);
  r.u[1] = *(const u32x4*)(p + 16);
  return r.f;
}

// B-matrix 32x16 bf16 fragment; source R row-major with R's rows == B's columns.
// lane<16: col=lane, K 0..7 & 16..23 ; lane>=16: col=lane-16, K 8..15 & 24..31
__device__ __forceinline__ bfrag load_b_frag(const unsigned short* base, int ld, int lane) {
  const int n    = lane & 15;
  const int koff = (lane < 16) ? 0 : 8;
  const unsigned short* p = base + (size_t)n * ld + koff;
  FragU r;
  r.u[0] = *(const u32x4*)(p);
  r.u[1] = *(const u32x4*)(p + 16);
  return r.f;
}

__device__ __forceinline__ v8f wmma_bf16(bfrag a, bfrag b, v8f c) {
  return __builtin_amdgcn_wmma_f32_16x16x32_bf16(false, a, false, b, (short)0, c, false, false);
}

#if USE_TDM
__device__ __forceinline__ unsigned int lds_off(const void* p) {
  // generic LDS address = {aperture, lds_byte_offset}; low 32 bits are the LDS offset
  return (unsigned int)(uintptr_t)p;
}

// 2D tensor-DMA: global (strided rows) -> LDS (dense). data_size = 2 bytes (bf16).
// Descriptor bit layout per CDNA5 ISA ch.8 (D# groups 0/1); groups 2/3 unused (2D).
__device__ __forceinline__ void tdm_load_2d(unsigned int lds_byte_off, const void* gaddr,
                                            unsigned int tile_d0, unsigned int tile_d1,
                                            unsigned int tensor_d0, unsigned int tensor_d1,
                                            unsigned long long stride0_elems) {
  unsigned long long ga = (unsigned long long)(uintptr_t)gaddr;
  u32x4 g0 = {0u, 0u, 0u, 0u};
  g0[0] = 1u;                                               // count=1, user descriptor
  g0[1] = lds_byte_off;                                     // lds_addr
  g0[2] = (unsigned int)(ga & 0xFFFFFFFFu);                 // global_addr[31:0]
  g0[3] = (unsigned int)((ga >> 32) & 0x1FFFFFFu) | (2u << 30);  // global_addr[56:32] | type=2
  i32x8 g1 = {0, 0, 0, 0, 0, 0, 0, 0};
  g1[0] = (int)(1u << 16);                                  // data_size=1 -> 2 bytes; mask=0
  g1[1] = (int)((tensor_d0 & 0xFFFFu) << 16);               // tensor_dim0[15:0] @ bits 63:48
  g1[2] = (int)(((tensor_d0 >> 16) & 0xFFFFu) | ((tensor_d1 & 0xFFFFu) << 16));
  g1[3] = (int)(((tensor_d1 >> 16) & 0xFFFFu) | ((tile_d0 & 0xFFFFu) << 16));
  g1[4] = (int)(tile_d1 & 0xFFFFu);                         // tile_dim1; tile_dim2=0
  g1[5] = (int)(unsigned int)(stride0_elems & 0xFFFFFFFFu); // tensor_dim0_stride[31:0]
  g1[6] = (int)(unsigned int)((stride0_elems >> 32) & 0xFFFFu);
  g1[7] = 0;
  i32x4 z4 = {0, 0, 0, 0};
#if defined(__clang_major__) && (__clang_major__ >= 23)
  i32x8 z8 = {0, 0, 0, 0, 0, 0, 0, 0};
  __builtin_amdgcn_tensor_load_to_lds(g0, g1, z4, z4, z8, 0);
#else
  __builtin_amdgcn_tensor_load_to_lds(g0, g1, z4, z4, 0);
#endif
}
#define TDM_WAIT(n) __builtin_amdgcn_s_wait_tensorcnt(n)
#else
// fallback: cooperative global->LDS copy (dense dst, strided src), 128 threads
__device__ __forceinline__ void coop_copy(unsigned short* dst, const unsigned short* src,
                                          int d0, int d1, int src_ld, int tid) {
  const int total = d0 * d1;
  for (int e = tid * 8; e < total; e += 128 * 8) {
    const int row = e / d0, col = e - row * d0;
    *(u32x4*)(dst + e) = *(const u32x4*)(src + (size_t)row * src_ld + col);
  }
}
#endif

// ---------------- prep kernels ----------------
__global__ void cvt_bf16_kernel(const float* __restrict__ in, unsigned short* __restrict__ out, int n) {
  int i = blockIdx.x * blockDim.x + threadIdx.x;
  if (i < n) out[i] = f2bf(in[i]);
}

__global__ void cvt_wt_kernel(const float* __restrict__ W, unsigned short* __restrict__ WT) {
  int i = blockIdx.x * blockDim.x + threadIdx.x;
  if (i < D_ * D_) {
    int n = i / D_, k = i - n * D_;
    WT[(size_t)n * D_ + k] = f2bf(W[(size_t)k * D_ + n]);
  }
}

// ---------------- projection GEMM: [BS,D] @ W + b ----------------
// X tile (64 x KB_) staged in LDS via TDM, double-buffered; W^T fragments software-pipelined.
// mode 0: Qh [B,H,S,HD] bf16; 1: Kh; 2: VhT [B,H,HD,S]; 3: f32 [BS,D]
__global__ __launch_bounds__(128) void proj_kernel(
    const unsigned short* __restrict__ X,
    const unsigned short* __restrict__ WT,
    const float* __restrict__ bias,
    unsigned short* __restrict__ out_bf,
    float* __restrict__ out_f,
    int mode)
{
  __shared__ unsigned short xs[2][64 * KB_];   // 2 x 16KB
  const int lane = threadIdx.x & 31;
  const int wave = threadIdx.x >> 5;
  const int hf   = lane >> 4;
  const int ln   = lane & 15;
  const int mblk = blockIdx.x * 64;
  const int n0   = blockIdx.y * 64;
  const unsigned short* Xblk = X + (size_t)mblk * D_;

#if USE_TDM
  if (threadIdx.x == 0)
    tdm_load_2d(lds_off(xs[0]), Xblk, KB_, 64, D_, BS_, D_);
#endif

  const v8f zacc = {0.f,0.f,0.f,0.f,0.f,0.f,0.f,0.f};
  v8f acc[4];
  #pragma unroll
  for (int t = 0; t < 4; ++t) acc[t] = zacc;

  // pipeline the weight-side B fragments (global, L2-resident)
  bfrag bcur[4];
  #pragma unroll
  for (int t = 0; t < 4; ++t)
    bcur[t] = load_b_frag(WT + (size_t)(n0 + t * 16) * D_, D_, lane);

  for (int kb = 0; kb < D_ / KB_; ++kb) {
    const int cur = kb & 1;
#if USE_TDM
    if (kb + 1 < D_ / KB_) {
      if (threadIdx.x == 0)
        tdm_load_2d(lds_off(xs[1 - cur]), Xblk + (size_t)(kb + 1) * KB_, KB_, 64, D_, BS_, D_);
      TDM_WAIT(1);               // current tile done; next DMA stays in flight
    } else {
      TDM_WAIT(0);
    }
#else
    coop_copy(xs[cur], Xblk + (size_t)kb * KB_, KB_, 64, D_, threadIdx.x);
#endif
    __syncthreads();

    #pragma unroll
    for (int kk = 0; kk < KB_; kk += 32) {
      bfrag a = load_a_frag(&xs[cur][(wave * 16) * KB_ + kk], KB_, lane);
      const int kgn = kb * KB_ + kk + 32;
      bfrag bn[4];
      if (kgn < D_) {            // prefetch next B fragments before issuing WMMAs
        #pragma unroll
        for (int t = 0; t < 4; ++t)
          bn[t] = load_b_frag(WT + (size_t)(n0 + t * 16) * D_ + kgn, D_, lane);
      }
      #pragma unroll
      for (int t = 0; t < 4; ++t) acc[t] = wmma_bf16(a, bcur[t], acc[t]);
      if (kgn < D_) {
        #pragma unroll
        for (int t = 0; t < 4; ++t) bcur[t] = bn[t];
      }
    }
    __syncthreads();             // protect LDS buffer before it is re-filled
  }

  #pragma unroll
  for (int t = 0; t < 4; ++t) {
    const int col = n0 + t * 16 + ln;
    const float bv = bias[col];
    #pragma unroll
    for (int r = 0; r < 8; ++r) {
      const float val = acc[t][r] + bv;
      const int row = mblk + wave * 16 + r + 8 * hf;
      if (mode == 3) {
        out_f[(size_t)row * D_ + col] = val;
      } else {
        const int bb = row >> 11, ss = row & (S_ - 1);
        const int hh = col >> 6,  hd = col & 63;
        size_t idx;
        if (mode == 2) idx = (((size_t)bb * H_ + hh) * HD_ + hd) * S_ + ss;
        else           idx = (((size_t)bb * H_ + hh) * S_  + ss) * HD_ + hd;
        out_bf[idx] = f2bf(val);
      }
    }
  }
}

// ---------------- flash attention (causal) ----------------
// grid (S/64, B*H), block 128. All 4 waves share the j-loop bound (== blockIdx.x),
// so K/V tiles are staged once per workgroup via double-buffered TDM.
__global__ __launch_bounds__(128) void attn_kernel(
    const unsigned short* __restrict__ Qh,   // [B,H,S,HD]
    const unsigned short* __restrict__ Kh,   // [B,H,S,HD]
    const unsigned short* __restrict__ VhT,  // [B,H,HD,S]
    unsigned short* __restrict__ attn_out)   // [B,S,D] bf16
{
  __shared__ unsigned short kls[2][64 * 64]; // 2 x 8KB key tiles
  __shared__ unsigned short vls[2][64 * 64]; // 2 x 8KB value tiles (hd-major)
  __shared__ unsigned short pls[4][16 * 64]; // per-wave P staging
  const int lane = threadIdx.x & 31;
  const int wave = threadIdx.x >> 5;
  const int hf   = lane >> 4;
  const int ln   = lane & 15;
  const int bh   = blockIdx.y;
  const int b    = bh / H_;
  const int h    = bh - b * H_;
  const int q0   = blockIdx.x * 64 + wave * 16;

  const unsigned short* qp = Qh  + ((size_t)bh * S_ + q0) * HD_;
  const unsigned short* kp = Kh  + (size_t)bh * S_ * HD_;
  const unsigned short* vp = VhT + (size_t)bh * HD_ * S_;
  unsigned short* wls = pls[wave];

  bfrag aq[2];
  aq[0] = load_a_frag(qp + 0,  HD_, lane);
  aq[1] = load_a_frag(qp + 32, HD_, lane);

  const v8f zacc = {0.f,0.f,0.f,0.f,0.f,0.f,0.f,0.f};
  v8f o[4];
  float m_i[8], l_i[8];
  #pragma unroll
  for (int t = 0; t < 4; ++t) o[t] = zacc;
  #pragma unroll
  for (int r = 0; r < 8; ++r) { m_i[r] = -1e30f; l_i[r] = 0.f; }

  const int jbMax = blockIdx.x;   // == (q0+15)>>6 for every wave in the block

#if USE_TDM
  if (threadIdx.x == 0) {
    tdm_load_2d(lds_off(kls[0]), kp, 4096, 1, 4096, 1, 4096);       // contiguous 8KB
    tdm_load_2d(lds_off(vls[0]), vp, 64, 64, S_, HD_, S_);          // 64 rows, stride S
  }
#endif

  for (int jb = 0; jb <= jbMax; ++jb) {
    const int jBase = jb * 64;
    const int cur = jb & 1;
#if USE_TDM
    if (jb < jbMax) {
      if (threadIdx.x == 0) {
        tdm_load_2d(lds_off(kls[1 - cur]), kp + (size_t)(jBase + 64) * HD_, 4096, 1, 4096, 1, 4096);
        tdm_load_2d(lds_off(vls[1 - cur]), vp + jBase + 64, 64, 64, S_, HD_, S_);
      }
      TDM_WAIT(2);               // this block's 2 tiles done; next 2 DMAs in flight
    } else {
      TDM_WAIT(0);
    }
#else
    coop_copy(kls[cur], kp + (size_t)jBase * HD_, 4096, 1, 4096, threadIdx.x);
    coop_copy(vls[cur], vp + jBase, 64, 64, S_, threadIdx.x);
#endif
    __syncthreads();

    // S = Q @ K^T from LDS key tile
    v8f st[4];
    #pragma unroll
    for (int t = 0; t < 4; ++t) {
      v8f a = zacc;
      #pragma unroll
      for (int c = 0; c < 2; ++c) {
        bfrag bk = load_b_frag(&kls[cur][(t * 16) * 64 + c * 32], 64, lane);
        a = wmma_bf16(aq[c], bk, a);
      }
      st[t] = a;
    }

    // scale + causal mask (C/D layout: row = r + 8*hf, col = t*16 + ln)
    #pragma unroll
    for (int t = 0; t < 4; ++t) {
      #pragma unroll
      for (int r = 0; r < 8; ++r) {
        float s = st[t][r] * 0.125f;
        const int kIdx = jBase + t * 16 + ln;
        const int qIdx = q0 + r + 8 * hf;
        st[t][r] = (kIdx > qIdx) ? (s - 1e9f) : s;
      }
    }

    // online softmax; rows span 16-lane halves
    #pragma unroll
    for (int r = 0; r < 8; ++r) {
      float vmax = fmaxf(fmaxf(st[0][r], st[1][r]), fmaxf(st[2][r], st[3][r]));
      #pragma unroll
      for (int sh = 1; sh < 16; sh <<= 1) vmax = fmaxf(vmax, __shfl_xor(vmax, sh, 32));
      const float mnew = fmaxf(m_i[r], vmax);
      const float corr = __expf(m_i[r] - mnew);
      m_i[r] = mnew;
      float rs = 0.f;
      #pragma unroll
      for (int t = 0; t < 4; ++t) {
        const float p = __expf(st[t][r] - mnew);
        st[t][r] = p;
        rs += p;
      }
      #pragma unroll
      for (int sh = 1; sh < 16; sh <<= 1) rs += __shfl_xor(rs, sh, 32);
      l_i[r] = l_i[r] * corr + rs;
      #pragma unroll
      for (int t = 0; t < 4; ++t) o[t][r] *= corr;
    }

    // reshape P -> A fragments via wave-private LDS (in-order per wave, no barrier)
    #pragma unroll
    for (int t = 0; t < 4; ++t) {
      #pragma unroll
      for (int r = 0; r < 8; ++r)
        wls[(r + 8 * hf) * 64 + t * 16 + ln] = f2bf(st[t][r]);
    }
    bfrag ap0 = load_a_frag(wls + 0,  64, lane);
    bfrag ap1 = load_a_frag(wls + 32, 64, lane);

    // O += P @ V from LDS value tile
    #pragma unroll
    for (int t = 0; t < 4; ++t) {
      bfrag bv0 = load_b_frag(&vls[cur][(t * 16) * 64 + 0],  64, lane);
      bfrag bv1 = load_b_frag(&vls[cur][(t * 16) * 64 + 32], 64, lane);
      o[t] = wmma_bf16(ap0, bv0, o[t]);
      o[t] = wmma_bf16(ap1, bv1, o[t]);
    }
    __syncthreads();             // all waves done with tile before next DMA overwrites
  }

  #pragma unroll
  for (int t = 0; t < 4; ++t) {
    const int col = h * HD_ + t * 16 + ln;
    #pragma unroll
    for (int r = 0; r < 8; ++r) {
      const int qIdx = q0 + r + 8 * hf;
      const float val = o[t][r] / l_i[r];
      attn_out[((size_t)b * S_ + qIdx) * D_ + col] = f2bf(val);
    }
  }
}

// ---------------- host launch ----------------
extern "C" void kernel_launch(void* const* d_in, const int* in_sizes, int n_in,
                              void* d_out, int out_size, void* d_ws, size_t ws_size,
                              hipStream_t stream) {
  (void)in_sizes; (void)n_in; (void)out_size; (void)ws_size;
  const float* q  = (const float*)d_in[0];
  const float* k  = (const float*)d_in[1];
  const float* v  = (const float*)d_in[2];
  // d_in[3] = causal mask (triu) -- computed analytically in-kernel
  const float* Wq = (const float*)d_in[4];
  const float* bq = (const float*)d_in[5];
  const float* Wk = (const float*)d_in[6];
  const float* bk = (const float*)d_in[7];
  const float* Wv = (const float*)d_in[8];
  const float* bv = (const float*)d_in[9];
  const float* Wo = (const float*)d_in[10];
  const float* bo = (const float*)d_in[11];
  float* out = (float*)d_out;

  char* ws = (char*)d_ws;
  size_t off = 0;
  auto alloc = [&](size_t bytes) -> char* {
    char* p = ws + off;
    off += (bytes + 255) & ~(size_t)255;
    return p;
  };
  unsigned short* Xq  = (unsigned short*)alloc((size_t)BS_ * D_ * 2);
  unsigned short* Xk  = (unsigned short*)alloc((size_t)BS_ * D_ * 2);
  unsigned short* Xv  = (unsigned short*)alloc((size_t)BS_ * D_ * 2);
  unsigned short* WqT = (unsigned short*)alloc((size_t)D_ * D_ * 2);
  unsigned short* WkT = (unsigned short*)alloc((size_t)D_ * D_ * 2);
  unsigned short* WvT = (unsigned short*)alloc((size_t)D_ * D_ * 2);
  unsigned short* WoT = (unsigned short*)alloc((size_t)D_ * D_ * 2);
  unsigned short* Qh  = (unsigned short*)alloc((size_t)B_ * H_ * S_ * HD_ * 2);
  unsigned short* Kh  = (unsigned short*)alloc((size_t)B_ * H_ * S_ * HD_ * 2);
  unsigned short* VhT = (unsigned short*)alloc((size_t)B_ * H_ * S_ * HD_ * 2);
  unsigned short* Att = (unsigned short*)alloc((size_t)BS_ * D_ * 2);

  const int nX = BS_ * D_;
  cvt_bf16_kernel<<<(nX + 255) / 256, 256, 0, stream>>>(q, Xq, nX);
  cvt_bf16_kernel<<<(nX + 255) / 256, 256, 0, stream>>>(k, Xk, nX);
  cvt_bf16_kernel<<<(nX + 255) / 256, 256, 0, stream>>>(v, Xv, nX);

  const int nW = D_ * D_;
  cvt_wt_kernel<<<(nW + 255) / 256, 256, 0, stream>>>(Wq, WqT);
  cvt_wt_kernel<<<(nW + 255) / 256, 256, 0, stream>>>(Wk, WkT);
  cvt_wt_kernel<<<(nW + 255) / 256, 256, 0, stream>>>(Wv, WvT);
  cvt_wt_kernel<<<(nW + 255) / 256, 256, 0, stream>>>(Wo, WoT);

  dim3 gp(BS_ / 64, D_ / 64);
  proj_kernel<<<gp, 128, 0, stream>>>(Xq, WqT, bq, Qh,  nullptr, 0);
  proj_kernel<<<gp, 128, 0, stream>>>(Xk, WkT, bk, Kh,  nullptr, 1);
  proj_kernel<<<gp, 128, 0, stream>>>(Xv, WvT, bv, VhT, nullptr, 2);

  dim3 ga(S_ / 64, B_ * H_);
  attn_kernel<<<ga, 128, 0, stream>>>(Qh, Kh, VhT, Att);

  proj_kernel<<<gp, 128, 0, stream>>>(Att, WoT, bo, nullptr, out, 3);
}
